// _AdapterAttention_52673478918510
// MI455X (gfx1250) — matrix operand order repaired and checked by reading, not compiled, not run on hardware
//
#include <hip/hip_runtime.h>
#include <stdint.h>

#define DM 2048
#define SQ 2048
#define HD 128
#define NH 16

typedef _Float16 h16v __attribute__((ext_vector_type(16)));
typedef _Float16 h8v  __attribute__((ext_vector_type(8)));
typedef float    f8v  __attribute__((ext_vector_type(8)));
typedef float    f4v  __attribute__((ext_vector_type(4)));

static __device__ __forceinline__ f8v wmma16(h16v a, h16v b, f8v c) {
  return __builtin_amdgcn_wmma_f32_16x16x32_f16(false, a, false, b, (short)0, c, false, false);
}

// A-matrix fragment (16x32 f16): lanes 0-15 rows M, halves K+{0..7,16..23};
// lanes 16-31 same rows, K+{8..15,24..31}.
static __device__ __forceinline__ h16v fragA(const _Float16* base, int ldh,
                                             int rowBase, int kBase, int lane) {
  const _Float16* p = base + (rowBase + (lane & 15)) * ldh + kBase + ((lane >> 4) << 3);
  h8v lo = *(const h8v*)(p);
  h8v hi = *(const h8v*)(p + 16);
  return __builtin_shufflevector(lo, hi, 0,1,2,3,4,5,6,7,8,9,10,11,12,13,14,15);
}

// B-matrix fragment (32x16 f16): lanes 0-15 col N, halves K+{0..7,16..23};
// lanes 16-31 col N, halves K+{8..15,24..31}. Source row-major [N][K].
static __device__ __forceinline__ h16v fragB(const _Float16* base, int ldh,
                                             int nBase, int kBase, int lane) {
  const _Float16* p = base + (nBase + (lane & 15)) * ldh + kBase + ((lane >> 4) << 3);
  h8v lo = *(const h8v*)(p);
  h8v hi = *(const h8v*)(p + 16);
  return __builtin_shufflevector(lo, hi, 0,1,2,3,4,5,6,7,8,9,10,11,12,13,14,15);
}

// Stage 128x64 fp32 tile -> f16 LDS (row stride 72 halves), 256 threads.
static __device__ __forceinline__ void stage_f32(const float* __restrict__ src,
                                                 long ld, _Float16* dst) {
  int t = threadIdx.x;
  int row = t >> 1;
  int c0 = (t & 1) * 32;
  const float* s = src + (long)row * ld + c0;
  _Float16* d = dst + row * 72 + c0;
#pragma unroll
  for (int i = 0; i < 4; ++i) {
    f4v v0 = *(const f4v*)(s + i * 8);
    f4v v1 = *(const f4v*)(s + i * 8 + 4);
    h8v hv;
    hv[0] = (_Float16)v0[0]; hv[1] = (_Float16)v0[1];
    hv[2] = (_Float16)v0[2]; hv[3] = (_Float16)v0[3];
    hv[4] = (_Float16)v1[0]; hv[5] = (_Float16)v1[1];
    hv[6] = (_Float16)v1[2]; hv[7] = (_Float16)v1[3];
    *(h8v*)(d + i * 8) = hv;
  }
}

// Stage 128x64 f16 tile -> LDS (row stride 72 halves).
static __device__ __forceinline__ void stage_f16(const _Float16* __restrict__ src,
                                                 long ld, _Float16* dst) {
  int t = threadIdx.x;
  int row = t >> 1;
  int c0 = (t & 1) * 32;
  const _Float16* s = src + (long)row * ld + c0;
  _Float16* d = dst + row * 72 + c0;
#pragma unroll
  for (int i = 0; i < 4; ++i) *(h8v*)(d + i * 8) = *(const h8v*)(s + i * 8);
}

// Async-stage a 128x128 f16 tile (row stride 136 halves in LDS) using
// CDNA5 global->LDS async DMA. 8 x b128 per thread, tracked by ASYNCcnt.
static __device__ __forceinline__ void stage_async(const _Float16* __restrict__ src,
                                                   long ld, uint32_t ldsBytes) {
  int t = threadIdx.x;
  int row = t >> 1;
  int c0 = (t & 1) * 64;
  const _Float16* s = src + (long)row * ld + c0;
  uint32_t d = ldsBytes + (uint32_t)(row * 136 + c0) * 2u;
#pragma unroll
  for (int i = 0; i < 8; ++i) {
    asm volatile("global_load_async_to_lds_b128 %0, %1, off"
                 :: "v"(d + i * 16),
                    "v"((unsigned long long)(uintptr_t)(s + i * 8))
                 : "memory");
  }
}

static __device__ __forceinline__ void wait_async16() {
  asm volatile("s_wait_asynccnt 0x10" ::: "memory");
}
static __device__ __forceinline__ void wait_async0() {
  asm volatile("s_wait_asynccnt 0x0" ::: "memory");
}

// -------------------- QKV projection GEMM (fused norm/rope/transpose) ------
// out[m,n] = sum_k X[m,k] * W[n,k]; tile 128x128 per block, N-tile == one head.
// mode 0: RMS-norm + RoPE epilogue, store f16 [b][h][s][d] (*qscale)
// mode 1: store f16 transposed   [b][h][d][s]   (for V)
__global__ __launch_bounds__(256)
void k_gemm_qkv(const float* __restrict__ X, const float* __restrict__ W,
                const float* __restrict__ cosT, const float* __restrict__ sinT,
                const float* __restrict__ normW,
                _Float16* __restrict__ out, int mode, float qscale) {
  extern __shared__ _Float16 smem[];
  // LDS map (halves): A buffers at 0 / 9216, B buffers at 18432 / 27648.
  float* tileF = (float*)smem;  // 128x132 fp32, reused after main loop

  const int lane = threadIdx.x & 31;
  const int wid  = threadIdx.x >> 5;
  const int m0 = blockIdx.y * 128;
  const int n0 = blockIdx.x * 128;
  const int wm = (wid & 3) * 32;
  const int wn = (wid >> 2) * 64;

  f8v acc[2][4];
#pragma unroll
  for (int mi = 0; mi < 2; ++mi)
#pragma unroll
    for (int ni = 0; ni < 4; ++ni) acc[mi][ni] = (f8v){};

  stage_f32(X + (long)m0 * DM, DM, smem);
  stage_f32(W + (long)n0 * DM, DM, smem + 18432);

  const long rowOffA = (long)m0 * DM + (long)(threadIdx.x >> 1) * DM;
  const long rowOffB = (long)n0 * DM + (long)(threadIdx.x >> 1) * DM;

  const int NK = DM / 64;
  for (int kt = 0; kt < NK; ++kt) {
    __syncthreads();
    if (kt + 2 < NK) {  // warm L2/WGP$ two tiles ahead
      __builtin_prefetch(X + rowOffA + (kt + 2) * 64, 0, 0);
      __builtin_prefetch(W + rowOffB + (kt + 2) * 64, 0, 0);
    }
    if (kt + 1 < NK) {
      int nb = (kt + 1) & 1;
      stage_f32(X + (long)m0 * DM + (kt + 1) * 64, DM, smem + nb * 9216);
      stage_f32(W + (long)n0 * DM + (kt + 1) * 64, DM, smem + 18432 + nb * 9216);
    }
    const _Float16* a = smem + (kt & 1) * 9216;
    const _Float16* b = smem + 18432 + (kt & 1) * 9216;
#pragma unroll
    for (int ks = 0; ks < 2; ++ks) {
      h16v af[2];
#pragma unroll
      for (int mi = 0; mi < 2; ++mi) af[mi] = fragA(a, 72, wm + mi * 16, ks * 32, lane);
#pragma unroll
      for (int ni = 0; ni < 4; ++ni) {
        h16v bf = fragB(b, 72, wn + ni * 16, ks * 32, lane);
#pragma unroll
        for (int mi = 0; mi < 2; ++mi) acc[mi][ni] = wmma16(af[mi], bf, acc[mi][ni]);
      }
    }
  }
  __syncthreads();

  // spill accumulators to LDS fp32 tile for the row-wise epilogue
#pragma unroll
  for (int mi = 0; mi < 2; ++mi)
#pragma unroll
    for (int ni = 0; ni < 4; ++ni)
#pragma unroll
      for (int r = 0; r < 8; ++r) {
        int row = wm + mi * 16 + ((lane >> 4) << 3) + r;
        int col = wn + ni * 16 + (lane & 15);
        tileF[row * 132 + col] = acc[mi][ni][r];
      }
  __syncthreads();

  const int bb = m0 >> 11;      // SQ == 2048
  const int s0 = m0 & (SQ - 1);
  const int h  = blockIdx.x;

  if (mode == 0) {
    // RMS norm over hd=128 (full tile width) + RoPE, then f16 store [b][h][s][d]
    int j  = threadIdx.x >> 1;
    int d0 = (threadIdx.x & 1) * 64;
    const float* rowp = tileF + j * 132;
    float ss = 0.f;
#pragma unroll 8
    for (int d = 0; d < HD; ++d) ss += rowp[d] * rowp[d];
    float inv = rsqrtf(ss * (1.0f / HD) + 1e-6f);
    int s = s0 + j;
    const float* cp = cosT + ((long)bb * SQ + s) * HD;
    const float* sp = sinT + ((long)bb * SQ + s) * HD;
    _Float16* op = out + (((long)(bb * NH + h) * SQ) + s) * HD;
    for (int dd = 0; dd < 64; ++dd) {
      int d  = d0 + dd;
      int pd = (d < 64) ? d + 64 : d - 64;
      float xn  = rowp[d]  * inv * normW[d];
      float xr  = rowp[pd] * inv * normW[pd];
      float rot = (d < 64) ? -xr : xr;
      op[d] = (_Float16)((xn * cp[d] + rot * sp[d]) * qscale);
    }
  } else {
    // V: store transposed f16 [b][h][d][s]
    int d  = threadIdx.x & 127;
    int r0 = (threadIdx.x >> 7) * 64;
    _Float16* op = out + ((long)(bb * NH + h) * HD + d) * SQ + s0 + r0;
    for (int r = 0; r < 64; ++r)
      op[r] = (_Float16)tileF[(r0 + r) * 132 + d];
  }
}

// -------------------- flash attention --------------------------------------
// Q [bh][s][d], K [bh][s][d], Vt [bh][d][s] (all f16); Y [b][s][h][d] f16.
// 1/sqrt(hd) folded into Q. K/V tiles double-buffered via async global->LDS
// DMA: tile kt+1 streams in (ASYNCcnt) while tile kt is consumed by WMMA.
__global__ __launch_bounds__(256)
void k_attn(const _Float16* __restrict__ Q, const _Float16* __restrict__ K,
            const _Float16* __restrict__ Vt, _Float16* __restrict__ Y) {
  extern __shared__ _Float16 smem[];
  // LDS map (halves): Qs @0, Ps @17408, K0 @34816, V0 @52224, K1 @69632, V1 @87040
  _Float16* Qs = smem;
  _Float16* Ps = smem + 17408;
  const uint32_t smemB = (uint32_t)(uintptr_t)(void*)smem;

  const int lane = threadIdx.x & 31;
  const int wid  = threadIdx.x >> 5;
  const int qt = blockIdx.x, h = blockIdx.y, b = blockIdx.z;
  const int bh = b * NH + h;

  const _Float16* Kg = K  + (long)bh * SQ * HD;   // [s][d]
  const _Float16* Vg = Vt + (long)bh * HD * SQ;   // [d][s]

  // Prologue: async-stage Q (8 ops) then K0/V0 (16 ops); wait<=16 => Q landed.
  stage_async(Q + ((long)bh * SQ + qt * 128) * HD, HD, smemB);
  stage_async(Kg, HD, smemB + 34816 * 2);
  stage_async(Vg, SQ, smemB + 52224 * 2);
  wait_async16();
  __syncthreads();

  const int j0 = wid * 16;
  h16v aQ[4];
#pragma unroll
  for (int ks = 0; ks < 4; ++ks) aQ[ks] = fragA(Qs, 136, j0, ks * 32, lane);

  f8v o[8];
#pragma unroll
  for (int nf = 0; nf < 8; ++nf) o[nf] = (f8v){};
  float mi_[8], li[8];
#pragma unroll
  for (int r = 0; r < 8; ++r) { mi_[r] = -3.0e38f; li[r] = 0.f; }

  const int NT = SQ / 128;
  for (int kt = 0; kt < NT; ++kt) {
    // stream next tile while computing this one
    if (kt + 1 < NT) {
      uint32_t kb = smemB + (34816 + ((kt + 1) & 1) * 34816) * 2;
      uint32_t vb = smemB + (52224 + ((kt + 1) & 1) * 34816) * 2;
      stage_async(Kg + (long)(kt + 1) * 128 * HD, HD, kb);
      stage_async(Vg + (kt + 1) * 128, SQ, vb);
      wait_async16();  // oldest 16 (tile kt) complete; tile kt+1 may be in flight
    } else {
      wait_async0();
    }
    __syncthreads();

    const _Float16* Ks = smem + 34816 + (kt & 1) * 34816;
    const _Float16* Vs = smem + 52224 + (kt & 1) * 34816;

    // S = Q @ K^T   (16 rows per wave x 128 keys)
    f8v sf[8];
#pragma unroll
    for (int nf = 0; nf < 8; ++nf) {
      sf[nf] = (f8v){};
#pragma unroll
      for (int ks = 0; ks < 4; ++ks)
        sf[nf] = wmma16(aQ[ks], fragB(Ks, 136, nf * 16, ks * 32, lane), sf[nf]);
    }

    // online softmax per row (row = j0 + r + 8*(lane>=16); cols across 16-lane half)
#pragma unroll
    for (int r = 0; r < 8; ++r) {
      float mx = sf[0][r];
#pragma unroll
      for (int nf = 1; nf < 8; ++nf) mx = fmaxf(mx, sf[nf][r]);
#pragma unroll
      for (int off = 1; off < 16; off <<= 1) mx = fmaxf(mx, __shfl_xor(mx, off, 32));
      float mnew  = fmaxf(mi_[r], mx);
      float alpha = __expf(mi_[r] - mnew);
      int row = j0 + r + ((lane >> 4) << 3);
      float rs = 0.f;
#pragma unroll
      for (int nf = 0; nf < 8; ++nf) {
        float p = __expf(sf[nf][r] - mnew);
        rs += p;
        Ps[row * 136 + nf * 16 + (lane & 15)] = (_Float16)p;
      }
#pragma unroll
      for (int off = 1; off < 16; off <<= 1) rs += __shfl_xor(rs, off, 32);
      li[r] = li[r] * alpha + rs;
      mi_[r] = mnew;
#pragma unroll
      for (int nf = 0; nf < 8; ++nf) o[nf][r] *= alpha;
    }
    __syncthreads();

    // O += P @ V   (B cols are Vt rows -> contiguous)
#pragma unroll
    for (int ks = 0; ks < 4; ++ks) {
      h16v aP = fragA(Ps, 136, j0, ks * 32, lane);
#pragma unroll
      for (int nf = 0; nf < 8; ++nf)
        o[nf] = wmma16(aP, fragB(Vs, 136, nf * 16, ks * 32, lane), o[nf]);
    }
    __syncthreads();  // all reads of buf[kt&1] done before it is re-staged
  }

  // normalize and store f16 to Y [b][s][h][d]
#pragma unroll
  for (int r = 0; r < 8; ++r) {
    float invl = 1.0f / li[r];
    int s = qt * 128 + j0 + ((lane >> 4) << 3) + r;
    long base = (((long)b * SQ + s) * NH + h) * HD + (lane & 15);
#pragma unroll
    for (int nf = 0; nf < 8; ++nf)
      Y[base + nf * 16] = (_Float16)(o[nf][r] * invl);
  }
}

// -------------------- output projection ------------------------------------
// Out[m,n] = sum_k Yb[m,k] * Wo[n,k]; fp32 direct store.
__global__ __launch_bounds__(256)
void k_gemm_out(const _Float16* __restrict__ Yb, const float* __restrict__ Wo,
                float* __restrict__ Out) {
  extern __shared__ _Float16 smem[];
  // LDS map (halves): A buffers at 0 / 9216, B buffers at 18432 / 27648.

  const int lane = threadIdx.x & 31;
  const int wid  = threadIdx.x >> 5;
  const int m0 = blockIdx.y * 128;
  const int n0 = blockIdx.x * 128;
  const int wm = (wid & 3) * 32;
  const int wn = (wid >> 2) * 64;

  f8v acc[2][4];
#pragma unroll
  for (int mi = 0; mi < 2; ++mi)
#pragma unroll
    for (int ni = 0; ni < 4; ++ni) acc[mi][ni] = (f8v){};

  stage_f16(Yb + (long)m0 * DM, DM, smem);
  stage_f32(Wo + (long)n0 * DM, DM, smem + 18432);

  const long rowOffA = (long)m0 * DM + (long)(threadIdx.x >> 1) * DM;
  const long rowOffB = (long)n0 * DM + (long)(threadIdx.x >> 1) * DM;

  const int NK = DM / 64;
  for (int kt = 0; kt < NK; ++kt) {
    __syncthreads();
    if (kt + 2 < NK) {
      __builtin_prefetch(Yb + rowOffA + (kt + 2) * 64, 0, 0);
      __builtin_prefetch(Wo + rowOffB + (kt + 2) * 64, 0, 0);
    }
    if (kt + 1 < NK) {
      int nb = (kt + 1) & 1;
      stage_f16(Yb + (long)m0 * DM + (kt + 1) * 64, DM, smem + nb * 9216);
      stage_f32(Wo + (long)n0 * DM + (kt + 1) * 64, DM, smem + 18432 + nb * 9216);
    }
    const _Float16* a = smem + (kt & 1) * 9216;
    const _Float16* b = smem + 18432 + (kt & 1) * 9216;
#pragma unroll
    for (int ks = 0; ks < 2; ++ks) {
      h16v af[2];
#pragma unroll
      for (int mi = 0; mi < 2; ++mi) af[mi] = fragA(a, 72, wm + mi * 16, ks * 32, lane);
#pragma unroll
      for (int ni = 0; ni < 4; ++ni) {
        h16v bf = fragB(b, 72, wn + ni * 16, ks * 32, lane);
#pragma unroll
        for (int mi = 0; mi < 2; ++mi) acc[mi][ni] = wmma16(af[mi], bf, acc[mi][ni]);
      }
    }
  }

#pragma unroll
  for (int mi = 0; mi < 2; ++mi)
#pragma unroll
    for (int ni = 0; ni < 4; ++ni)
#pragma unroll
      for (int r = 0; r < 8; ++r) {
        int row = m0 + wm + mi * 16 + ((lane >> 4) << 3) + r;
        int col = n0 + wn + ni * 16 + (lane & 15);
        Out[(long)row * DM + col] = acc[mi][ni][r];
      }
}

// ---------------------------------------------------------------------------
extern "C" void kernel_launch(void* const* d_in, const int* in_sizes, int n_in,
                              void* d_out, int out_size, void* d_ws, size_t ws_size,
                              hipStream_t stream) {
  (void)in_sizes; (void)n_in; (void)out_size; (void)ws_size;
  const float* x      = (const float*)d_in[0];
  const float* ctx    = (const float*)d_in[1];
  const float* cos_q  = (const float*)d_in[2];
  const float* sin_q  = (const float*)d_in[3];
  const float* cos_k  = (const float*)d_in[4];
  const float* sin_k  = (const float*)d_in[5];
  const float* Wq     = (const float*)d_in[6];
  const float* Wk     = (const float*)d_in[7];
  const float* Wv     = (const float*)d_in[8];
  const float* Wo     = (const float*)d_in[9];
  const float* qnw    = (const float*)d_in[10];
  const float* knw    = (const float*)d_in[11];

  const long NTOK = 2L * SQ * DM;  // 8,388,608 halves per buffer
  _Float16* qb = (_Float16*)d_ws;
  _Float16* kb = qb + NTOK;
  _Float16* vt = kb + NTOK;
  _Float16* yb = vt + NTOK;

  dim3 blk(256);
  dim3 g1(DM / 128, 2 * SQ / 128);       // 16 x 32
  const size_t ldsG = 73728;             // 4 staging buffers (or 128x132 fp32 tile)
  const float qscale = 0.08838834764831845f;  // 1/sqrt(128)

  k_gemm_qkv<<<g1, blk, ldsG, stream>>>(x,   Wq, cos_q, sin_q, qnw, qb, 0, qscale);
  k_gemm_qkv<<<g1, blk, ldsG, stream>>>(ctx, Wk, cos_k, sin_k, knw, kb, 0, 1.0f);
  k_gemm_qkv<<<g1, blk, ldsG, stream>>>(ctx, Wv, nullptr, nullptr, nullptr, vt, 1, 1.0f);

  dim3 ga(SQ / 128, NH, 2);              // 16 x 16 x 2
  const size_t ldsA = 104448 * 2;        // Q + P + 2x(K,V) tiles = 208896 B
  k_attn<<<ga, blk, ldsA, stream>>>(qb, kb, vt, yb);

  k_gemm_out<<<g1, blk, ldsG, stream>>>(yb, Wo, (float*)d_out);
}
